// MultiModalFusion_15152644620587
// MI455X (gfx1250) — matrix operand order repaired and checked by reading, not compiled, not run on hardware
//
#include <hip/hip_runtime.h>

// ---------------------------------------------------------------------------
// MultiModalFusion for MI455X (gfx1250, wave32, WMMA).
// B=32, M=2, C=256, H=W=16, N=256, K_im2col = 2304, tap-major K ordering.
// ---------------------------------------------------------------------------

typedef __attribute__((ext_vector_type(16))) _Float16 v16h;
typedef __attribute__((ext_vector_type(8)))  _Float16 v8h;
typedef __attribute__((ext_vector_type(8)))  float    v8f;
typedef __attribute__((ext_vector_type(4)))  float    v4f;

#define BB   32
#define CC   256
#define NPIX 256
#define KTOT 2304   // 9 taps * 256 ic  (tap-major)
#define RATIO_ 0.8f
#define THR_   0.1f

#define LDSTRIDE 40          // halfs per panel row: 80B, 16B-aligned, conflict-free

// ---------------- workspace layout (bytes) ----------------
constexpr size_t OFF_MID    = 0;                                   // 64*256*256 f32
constexpr size_t OFF_XREC   = OFF_MID    + 64ull*256*256*4;
constexpr size_t OFF_REC    = OFF_XREC   + 64ull*256*256*4;        // 32*256*256 f32
constexpr size_t OFF_REC2   = OFF_REC    + 32ull*256*256*4;
constexpr size_t OFF_A16    = OFF_REC2   + 32ull*256*256*4;        // 32*256*256 f16
constexpr size_t OFF_BT16   = OFF_A16    + 32ull*256*256*2;
constexpr size_t OFF_CNT    = OFF_BT16   + 32ull*256*256*2;        // 64 int
constexpr size_t OFF_SCORES = OFF_CNT    + 256;                    // 32*2*256 f32
constexpr size_t OFF_LPART  = OFF_SCORES + 32ull*2*256*4;          // 1024 f32
constexpr size_t OFF_MMN    = OFF_LPART  + 1024*4;                 // 512 f32
constexpr size_t OFF_MMX    = OFF_MMN    + 512*4;                  // 512 f32
constexpr size_t OFF_MM     = OFF_MMX    + 512*4;                  // 2 f32 (+pad)
constexpr size_t OFF_WENC16 = OFF_MM     + 16;                     // 256*2304 f16
constexpr size_t OFF_WDEC16 = OFF_WENC16 + 256ull*2304*2;          // 256*2304 f16

// ---------------------------------------------------------------------------
// 0) one-time weight repack: w(oc, ic, tap) f32 -> w16(oc, tap, ic) f16
// ---------------------------------------------------------------------------
__global__ __launch_bounds__(256) void repack_w(const float* __restrict__ w,
                                                _Float16* __restrict__ w16)
{
    int idx = blockIdx.x * 256 + threadIdx.x;   // 589824 total
    int oc  = idx / KTOT;
    int r   = idx - oc * KTOT;
    int tap = r >> 8;
    int ic  = r & 255;
    w16[idx] = (_Float16)w[oc * KTOT + ic * 9 + tap];
}

// ---------------------------------------------------------------------------
// 1) data-dependent rank budget (counts)
// ---------------------------------------------------------------------------
__global__ void count_kernel(const float* __restrict__ feat_v,
                             int* __restrict__ counts)
{
    int b = threadIdx.x;
    if (b >= BB) return;
    int c0 = 0, c1 = 0;
    for (int i = 0; i < CC; ++i) {
        c0 += (feat_v[(b * 2 + 0) * CC + i] > THR_);
        c1 += (feat_v[(b * 2 + 1) * CC + i] > THR_);
    }
    int total = c0 + c1;                       // kmin = DIM = 256
    if (total > 256) {
        float t = (float)total;
        int a0  = (int)rintf(256.0f * (float)c0 / t);   // jnp.round: half-even
        int a1  = 256 - a0;                              // adj[:,-1] = DIM - adj[:,0]
        c0 = a0 > 0 ? a0 : 0;
        c1 = a1 > 0 ? a1 : 0;
    }
    counts[b * 2 + 0] = c0;
    counts[b * 2 + 1] = c1;
}

// ---------------------------------------------------------------------------
// 2) 3x3 SAME conv (256->256) as implicit GEMM on WMMA, tap-major K.
//    Double-buffered LDS panels, register-prefetched staging, b128 traffic.
// ---------------------------------------------------------------------------
__global__ __launch_bounds__(256) void conv3x3_wmma(
    const float*    __restrict__ x,      // (G,256,16,16) f32
    const _Float16* __restrict__ w16,    // (256, 9, 256) f16, tap-major
    const float*    __restrict__ bias,   // (256)
    float*          __restrict__ y,      // (G,256,256) f32
    int do_relu)
{
    __shared__ __align__(16) _Float16 ldsA[2][64 * LDSTRIDE];  // [oc][k]
    __shared__ __align__(16) _Float16 ldsB[2][64 * LDSTRIDE];  // [px][k]

    const int tid  = threadIdx.x;
    const int img  = blockIdx.x >> 4;
    const int tile = blockIdx.x & 15;
    const int oc0  = (tile >> 2) * 64;
    const int p0   = (tile & 3) * 64;
    const int wave = tid >> 5;
    const int lane = tid & 31;
    const int m    = lane & 15;
    const int hi   = lane >> 4;

    const int pxSub  = wave & 3;
    const int ocSub0 = wave >> 2;        // 0..1
    const int ocSub1 = ocSub0 + 2;       // 2..3

    // staging geometry (constant per thread)
    const int ai  = tid >> 2;            // A row (oc in tile) 0..63
    const int ag  = tid & 3;             // A 8-half group
    const int bp  = tid & 63;            // B pixel 0..63
    const int bk  = (tid >> 6) * 8;      // B k-base: 8 consecutive k
    const int py  = (p0 + bp) >> 4;
    const int pxx = (p0 + bp) & 15;

    const _Float16* wrow = w16 + (size_t)(oc0 + ai) * KTOT + ag * 8;
    const float*    ximg = x + (size_t)img * CC * NPIX;

    v8h   ra;           // A staging regs (8 halfs = 16B)
    float rb[8];        // B staging regs

    auto stage = [&](int kt) {
        const int tap = kt >> 3;              // scalar per iteration
        const int icb = (kt & 7) << 5;
        const int dy  = tap / 3 - 1;
        const int dx  = tap - (tap / 3) * 3 - 1;
        ra = *(const v8h*)(wrow + tap * 256 + icb);
        const int yy  = py  + dy;
        const int xx  = pxx + dx;
        const bool inb = ((unsigned)yy < 16u) & ((unsigned)xx < 16u);
        const int sp  = ((yy & 15) << 4) | (xx & 15);   // masked -> any valid addr ok
        const float* xc = ximg + (size_t)(icb + bk) * NPIX + sp;
        #pragma unroll
        for (int e = 0; e < 8; ++e) rb[e] = xc[(size_t)e * NPIX];
        #pragma unroll
        for (int e = 0; e < 8; ++e) rb[e] = inb ? rb[e] : 0.0f;   // cndmask
    };
    auto commit = [&](int buf) {
        *(v8h*)&ldsA[buf][ai * LDSTRIDE + ag * 8] = ra;           // ds_store_b128
        #pragma unroll
        for (int j = 0; j < 4; ++j) {                             // 4x ds_store_b32
            union { _Float16 h[2]; unsigned u; } pk;
            pk.h[0] = (_Float16)rb[2 * j];
            pk.h[1] = (_Float16)rb[2 * j + 1];
            *(unsigned*)&ldsB[buf][bp * LDSTRIDE + bk + 2 * j] = pk.u;
        }
    };

    v8f acc0 = {};
    v8f acc1 = {};

    stage(0);
    for (int kt = 0; kt < KTOT / 32; ++kt) {
        const int buf = kt & 1;
        commit(buf);
        __syncthreads();                               // one barrier per K-step
        if (kt < KTOT / 32 - 1) stage(kt + 1);         // overlap next loads

        union F { v16h v; v4f q[2]; } a0f, a1f, bff;
        const _Float16* Ar0 = &ldsA[buf][(ocSub0 * 16 + m) * LDSTRIDE];
        const _Float16* Ar1 = &ldsA[buf][(ocSub1 * 16 + m) * LDSTRIDE];
        const _Float16* Br  = &ldsB[buf][(pxSub  * 16 + m) * LDSTRIDE];
        // A frag: halfs [hi*8 .. +7] and [16+hi*8 .. +7]  (K = (h&7)+(h>>3)*16+hi*8)
        a0f.q[0] = *(const v4f*)(Ar0 + hi * 8);
        a0f.q[1] = *(const v4f*)(Ar0 + 16 + hi * 8);
        a1f.q[0] = *(const v4f*)(Ar1 + hi * 8);
        a1f.q[1] = *(const v4f*)(Ar1 + 16 + hi * 8);
        // B frag: halfs [hi*8 .. +7] and [16+hi*8 .. +7]  (K = (h&7)+(h>>3)*16+hi*8)
        bff.q[0] = *(const v4f*)(Br + hi * 8);
        bff.q[1] = *(const v4f*)(Br + 16 + hi * 8);

        acc0 = __builtin_amdgcn_wmma_f32_16x16x32_f16(false, a0f.v, false, bff.v,
                                                      (short)0, acc0, false, false);
        acc1 = __builtin_amdgcn_wmma_f32_16x16x32_f16(false, a1f.v, false, bff.v,
                                                      (short)0, acc1, false, false);
    }

    // ---- epilogue: bias (+ReLU); D: vgpr r -> row r+8*hi, lane -> col ----
    const int opx = p0 + pxSub * 16 + m;
    #pragma unroll
    for (int r = 0; r < 8; ++r) {
        int oc = oc0 + ocSub0 * 16 + r + 8 * hi;
        float v = acc0[r] + bias[oc];
        if (do_relu) v = fmaxf(v, 0.0f);
        y[((size_t)img * CC + oc) * NPIX + opx] = v;

        oc = oc0 + ocSub1 * 16 + r + 8 * hi;
        v  = acc1[r] + bias[oc];
        if (do_relu) v = fmaxf(v, 0.0f);
        y[((size_t)img * CC + oc) * NPIX + opx] = v;
    }
}

// ---------------------------------------------------------------------------
// 3) scatter-assemble:  A16[b][i][k] = valid_i ? fs[b,m_i,r_i,k] * v_k : 0
//                       BT16[b][j][k] = valid_j ? fd[b,m_j,k,r_j]      : 0
// ---------------------------------------------------------------------------
__global__ __launch_bounds__(256) void assemble_kernel(
    const float* __restrict__ feat_s,
    const float* __restrict__ feat_v,
    const float* __restrict__ feat_d,
    const int*   __restrict__ counts,
    _Float16* __restrict__ A16,
    _Float16* __restrict__ BT16)
{
    __shared__ float vv[256];
    __shared__ int   mr[256];           // m*256+r, or -1 for empty slot

    const int b     = blockIdx.x >> 3;
    const int chunk = blockIdx.x & 7;   // 32 rows per block
    const int tid   = threadIdx.x;
    const int c0    = counts[b * 2 + 0];
    const int c1    = counts[b * 2 + 1];

    {
        int d = tid, mi = -1, ri = 0;
        if (d < c0)            { mi = 0; ri = d; }
        else if (d < c0 + c1)  { mi = 1; ri = d - c0; }
        float v = 0.0f;
        if (mi >= 0) v = feat_v[(b * 2 + mi) * CC + ri];
        vv[d] = v;
        mr[d] = (mi >= 0) ? (mi * 256 + ri) : -1;
    }
    __syncthreads();

    const float myv = vv[tid];          // v_k for k == tid
    #pragma unroll 4
    for (int e0 = 0; e0 < 32; ++e0) {
        int e  = chunk * 32 + e0;       // slot (row i / col j)
        int me = mr[e];
        float a = 0.0f, d = 0.0f;
        if (me >= 0) {
            int mi = me >> 8, ri = me & 255;
            a = feat_s[((size_t)(b * 2 + mi) * CC + ri) * CC + tid] * myv;
            d = feat_d[((size_t)(b * 2 + mi) * NPIX + tid) * NPIX + ri];
        }
        A16 [((size_t)b * 256 + e) * 256 + tid] = (_Float16)a;
        BT16[((size_t)b * 256 + e) * 256 + tid] = (_Float16)d;
    }
}

// ---------------------------------------------------------------------------
// 4) rec[b] = A16[b] (256x256) * BT16[b]^T; one wave per 16x16 tile, b128 frags
// ---------------------------------------------------------------------------
__global__ __launch_bounds__(256) void gemm_rec_wmma(
    const _Float16* __restrict__ A16,
    const _Float16* __restrict__ BT16,
    float* __restrict__ rec)
{
    const int wave   = threadIdx.x >> 5;
    const int lane   = threadIdx.x & 31;
    const int tileId = blockIdx.x * 8 + wave;
    const int b      = tileId >> 8;
    const int t      = tileId & 255;
    const int i0     = (t >> 4) << 4;
    const int j0     = (t & 15) << 4;
    const int m      = lane & 15;
    const int hi     = lane >> 4;

    const _Float16* Abase = A16  + ((size_t)b * 256 + i0 + m) * 256;
    const _Float16* Bbase = BT16 + ((size_t)b * 256 + j0 + m) * 256;

    v8f acc = {};
    #pragma unroll
    for (int kk = 0; kk < 8; ++kk) {
        union F { v16h v; v4f q[2]; } a, bf;
        a.q[0]  = *(const v4f*)(Abase + kk * 32 + hi * 8);
        a.q[1]  = *(const v4f*)(Abase + kk * 32 + 16 + hi * 8);
        bf.q[0] = *(const v4f*)(Bbase + kk * 32 + hi * 8);
        bf.q[1] = *(const v4f*)(Bbase + kk * 32 + 16 + hi * 8);
        acc = __builtin_amdgcn_wmma_f32_16x16x32_f16(false, a.v, false, bf.v,
                                                     (short)0, acc, false, false);
    }
    #pragma unroll
    for (int r = 0; r < 8; ++r)
        rec[((size_t)b * 256 + i0 + r + 8 * hi) * 256 + j0 + m] = acc[r];
}

// ---------------------------------------------------------------------------
// 5) reductions: MSE loss and global min/max (deterministic trees)
// ---------------------------------------------------------------------------
__global__ __launch_bounds__(256) void sqdiff_partial(
    const float* __restrict__ a, const float* __restrict__ b,
    float* __restrict__ partial, int n)
{
    __shared__ float red[256];
    float s = 0.0f;
    for (int i = blockIdx.x * 256 + threadIdx.x; i < n; i += gridDim.x * 256) {
        float d = a[i] - b[i];
        s += d * d;
    }
    red[threadIdx.x] = s; __syncthreads();
    for (int o = 128; o > 0; o >>= 1) {
        if (threadIdx.x < o) red[threadIdx.x] += red[threadIdx.x + o];
        __syncthreads();
    }
    if (threadIdx.x == 0) partial[blockIdx.x] = red[0];
}

__global__ __launch_bounds__(256) void loss_final(
    const float* __restrict__ partial, int np, float* __restrict__ out, float invn)
{
    __shared__ float red[256];
    float s = 0.0f;
    for (int i = threadIdx.x; i < np; i += 256) s += partial[i];
    red[threadIdx.x] = s; __syncthreads();
    for (int o = 128; o > 0; o >>= 1) {
        if (threadIdx.x < o) red[threadIdx.x] += red[threadIdx.x + o];
        __syncthreads();
    }
    if (threadIdx.x == 0) *out = red[0] * invn;
}

__global__ __launch_bounds__(256) void minmax_partial(
    const float* __restrict__ x, int n,
    float* __restrict__ pmn, float* __restrict__ pmx)
{
    __shared__ float smn[256], smx[256];
    float mn = 3.402823e38f, mx = -3.402823e38f;
    for (int i = blockIdx.x * 256 + threadIdx.x; i < n; i += gridDim.x * 256) {
        float v = x[i];
        mn = fminf(mn, v);
        mx = fmaxf(mx, v);
    }
    smn[threadIdx.x] = mn; smx[threadIdx.x] = mx; __syncthreads();
    for (int o = 128; o > 0; o >>= 1) {
        if (threadIdx.x < o) {
            smn[threadIdx.x] = fminf(smn[threadIdx.x], smn[threadIdx.x + o]);
            smx[threadIdx.x] = fmaxf(smx[threadIdx.x], smx[threadIdx.x + o]);
        }
        __syncthreads();
    }
    if (threadIdx.x == 0) { pmn[blockIdx.x] = smn[0]; pmx[blockIdx.x] = smx[0]; }
}

__global__ __launch_bounds__(256) void minmax_final(
    const float* __restrict__ pmn, const float* __restrict__ pmx,
    int np, float* __restrict__ mm)
{
    __shared__ float smn[256], smx[256];
    float mn = 3.402823e38f, mx = -3.402823e38f;
    for (int i = threadIdx.x; i < np; i += 256) {
        mn = fminf(mn, pmn[i]);
        mx = fmaxf(mx, pmx[i]);
    }
    smn[threadIdx.x] = mn; smx[threadIdx.x] = mx; __syncthreads();
    for (int o = 128; o > 0; o >>= 1) {
        if (threadIdx.x < o) {
            smn[threadIdx.x] = fminf(smn[threadIdx.x], smn[threadIdx.x + o]);
            smx[threadIdx.x] = fmaxf(smx[threadIdx.x], smx[threadIdx.x + o]);
        }
        __syncthreads();
    }
    if (threadIdx.x == 0) { mm[0] = smn[0]; mm[1] = smx[0]; }
}

// ---------------------------------------------------------------------------
// 6) value conv (512ch -> 1ch, 3x3) + sigmoid(relu) + softmax over modality
// ---------------------------------------------------------------------------
__global__ __launch_bounds__(256) void scores_kernel(
    const float* __restrict__ rec2,      // (32,256,16,16)
    const float* __restrict__ con_feat,  // (32,2,256,16,16)
    const float* __restrict__ w_val,     // (512*9)
    const float* __restrict__ b_val,     // (1)
    const float* __restrict__ mm,        // {min, max}
    float* __restrict__ scores)          // (32,2,256)
{
    __shared__ float wv[512 * 9];
    const int b   = blockIdx.x;
    const int tid = threadIdx.x;
    for (int i = tid; i < 512 * 9; i += 256) wv[i] = w_val[i];
    __syncthreads();

    const float mn = mm[0], mx = mm[1];
    const float scale = 2.0f / (mx - mn);
    const float off   = -mn * scale - 1.0f;

    const int py = tid >> 4, px = tid & 15;
    const float* recb = rec2     + (size_t)b * CC * NPIX;
    const float* cf0  = con_feat + (size_t)(b * 2 + 0) * CC * NPIX;
    const float* cf1  = con_feat + (size_t)(b * 2 + 1) * CC * NPIX;

    float accR = 0.0f, acc0 = 0.0f, acc1 = 0.0f;
    #pragma unroll 2
    for (int ch = 0; ch < CC; ++ch) {
        #pragma unroll
        for (int kh = 0; kh < 3; ++kh) {
            int yy = py + kh - 1;
            if ((unsigned)yy >= 16u) continue;
            #pragma unroll
            for (int kw = 0; kw < 3; ++kw) {
                int xx = px + kw - 1;
                if ((unsigned)xx >= 16u) continue;
                int tap = kh * 3 + kw;
                int sp  = yy * 16 + xx;
                accR += wv[ch * 9 + tap] * (recb[ch * NPIX + sp] * scale + off);
                float wm = wv[(256 + ch) * 9 + tap];
                acc0 += wm * cf0[ch * NPIX + sp];
                acc1 += wm * cf1[ch * NPIX + sp];
            }
        }
    }
    const float bv = b_val[0];
    float v0 = fmaxf(accR + acc0 + bv, 0.0f);
    float v1 = fmaxf(accR + acc1 + bv, 0.0f);
    float s0 = 1.0f / (1.0f + expf(-v0));
    float s1 = 1.0f / (1.0f + expf(-v1));
    float e0 = expf(s0), e1 = expf(s1);
    float inv = 1.0f / (e0 + e1);
    scores[(b * 2 + 0) * NPIX + tid] = e0 * inv;
    scores[(b * 2 + 1) * NPIX + tid] = e1 * inv;
}

// ---------------------------------------------------------------------------
// 7) fusion: out = 0.2*(s0*cf0 + s1*cf1) + 0.8*rec_norm
// ---------------------------------------------------------------------------
__global__ __launch_bounds__(256) void fuse_kernel(
    const float* __restrict__ con_feat,
    const float* __restrict__ rec2,
    const float* __restrict__ scores,
    const float* __restrict__ mm,
    float* __restrict__ out)
{
    const size_t idx = (size_t)blockIdx.x * 256 + threadIdx.x;   // < 2097152
    const int b = (int)(idx >> 16);
    const int c = (int)((idx >> 8) & 255);
    const int p = (int)(idx & 255);

    const float mn = mm[0], mx = mm[1];
    const float scale = 2.0f / (mx - mn);
    const float off   = -mn * scale - 1.0f;

    float rn = rec2[idx] * scale + off;
    float s0 = scores[(b * 2 + 0) * NPIX + p];
    float s1 = scores[(b * 2 + 1) * NPIX + p];
    float f0 = con_feat[((size_t)(b * 2 + 0) * CC + c) * NPIX + p];
    float f1 = con_feat[((size_t)(b * 2 + 1) * CC + c) * NPIX + p];
    out[idx] = (1.0f - RATIO_) * (s0 * f0 + s1 * f1) + RATIO_ * rn;
}

// ---------------------------------------------------------------------------
// launcher
// ---------------------------------------------------------------------------
extern "C" void kernel_launch(void* const* d_in, const int* in_sizes, int n_in,
                              void* d_out, int out_size, void* d_ws, size_t ws_size,
                              hipStream_t stream)
{
    (void)in_sizes; (void)n_in; (void)out_size; (void)ws_size;

    const float* con_feat = (const float*)d_in[0];
    const float* feat_s   = (const float*)d_in[1];
    const float* feat_v   = (const float*)d_in[2];
    const float* feat_d   = (const float*)d_in[3];
    const float* w_enc    = (const float*)d_in[4];
    const float* b_enc    = (const float*)d_in[5];
    const float* w_dec    = (const float*)d_in[6];
    const float* b_dec    = (const float*)d_in[7];
    const float* w_val    = (const float*)d_in[8];
    const float* b_val    = (const float*)d_in[9];

    char* ws = (char*)d_ws;
    float*    mid    = (float*)   (ws + OFF_MID);
    float*    xrec   = (float*)   (ws + OFF_XREC);
    float*    rec    = (float*)   (ws + OFF_REC);
    float*    rec2   = (float*)   (ws + OFF_REC2);
    _Float16* A16    = (_Float16*)(ws + OFF_A16);
    _Float16* BT16   = (_Float16*)(ws + OFF_BT16);
    int*      counts = (int*)     (ws + OFF_CNT);
    float*    scores = (float*)   (ws + OFF_SCORES);
    float*    lpart  = (float*)   (ws + OFF_LPART);
    float*    pmn    = (float*)   (ws + OFF_MMN);
    float*    pmx    = (float*)   (ws + OFF_MMX);
    float*    mm     = (float*)   (ws + OFF_MM);
    _Float16* wenc16 = (_Float16*)(ws + OFF_WENC16);
    _Float16* wdec16 = (_Float16*)(ws + OFF_WDEC16);

    float* fused = (float*)d_out;                       // 32*256*256
    float* loss  = ((float*)d_out) + 32ull * 256 * 256; // scalar after fused

    // one-time weight repack to tap-major f16
    repack_w<<<KTOT, 256, 0, stream>>>(w_enc, wenc16);
    repack_w<<<KTOT, 256, 0, stream>>>(w_dec, wdec16);

    // counts / rank budget
    count_kernel<<<1, 32, 0, stream>>>(feat_v, counts);

    // autoencoder branch: mid = relu(conv_enc(x)); x_rec = conv_dec(mid)
    conv3x3_wmma<<<64 * 16, 256, 0, stream>>>(con_feat, wenc16, b_enc, mid, 1);
    conv3x3_wmma<<<64 * 16, 256, 0, stream>>>(mid, wdec16, b_dec, xrec, 0);
    sqdiff_partial<<<1024, 256, 0, stream>>>(con_feat, xrec, lpart, 64 * 256 * 256);
    loss_final<<<1, 256, 0, stream>>>(lpart, 1024, loss, 1.0f / (64.0f * 256.0f * 256.0f));

    // scatter-assemble + bmm + decoder conv
    assemble_kernel<<<256, 256, 0, stream>>>(feat_s, feat_v, feat_d, counts, A16, BT16);
    gemm_rec_wmma<<<1024, 256, 0, stream>>>(A16, BT16, rec);
    conv3x3_wmma<<<32 * 16, 256, 0, stream>>>(rec, wdec16, b_dec, rec2, 0);

    // global min/max for normalization
    minmax_partial<<<512, 256, 0, stream>>>(rec2, 32 * 256 * 256, pmn, pmx);
    minmax_final<<<1, 256, 0, stream>>>(pmn, pmx, 512, mm);

    // value scores + fusion
    scores_kernel<<<32, 256, 0, stream>>>(rec2, con_feat, w_val, b_val, mm, scores);
    fuse_kernel<<<8192, 256, 0, stream>>>(con_feat, rec2, scores, mm, fused);
}
